// AdaptiveMoELayer_49383533969484
// MI455X (gfx1250) — matrix operand order repaired and checked by reading, compiled, not run on hardware
//
#include <hip/hip_runtime.h>
#include <math.h>

// Problem constants (match reference).
#define TB 2
#define TS 1024
#define TT (TB * TS)   // 2048 tokens
#define DD 1024        // hidden dim
#define FF 4096        // ffn dim
#define EE 8           // experts
#define CHUNK_F 512    // F-chunk held in LDS between the two GEMMs

typedef __bf16 bf16;
typedef __attribute__((ext_vector_type(16))) __bf16 bf16x16;
typedef __attribute__((ext_vector_type(8)))  __bf16 bf16x8;
typedef __attribute__((ext_vector_type(4)))  __bf16 bf16x4;
typedef __attribute__((ext_vector_type(8)))  float  f32x8;
typedef __attribute__((ext_vector_type(4)))  float  f32x4;

__device__ __forceinline__ f32x8 zero8() {
  f32x8 z = {0.f, 0.f, 0.f, 0.f, 0.f, 0.f, 0.f, 0.f};
  return z;
}

// ---------------------------------------------------------------------------
// WMMA bf16 fragment loaders (CDNA5 ISA 7.12.2 layouts, wave32).
// A-matrix 16x32: lane half 'hi' selects K base (0 or 8); halves 0..7 hold
// K = kofs..kofs+7, halves 8..15 hold K = kofs+16..kofs+23.
// (Compiler merges these into two b128 loads per fragment.)
// ---------------------------------------------------------------------------
__device__ __forceinline__ bf16x16 load_a_k32(const bf16* __restrict__ p, int kofs) {
  bf16x4 q0 = *(const bf16x4*)(p + kofs + 0);
  bf16x4 q1 = *(const bf16x4*)(p + kofs + 4);
  bf16x4 q2 = *(const bf16x4*)(p + kofs + 16);
  bf16x4 q3 = *(const bf16x4*)(p + kofs + 20);
  bf16x16 r;
  r[0]  = q0[0]; r[1]  = q0[1]; r[2]  = q0[2]; r[3]  = q0[3];
  r[4]  = q1[0]; r[5]  = q1[1]; r[6]  = q1[2]; r[7]  = q1[3];
  r[8]  = q2[0]; r[9]  = q2[1]; r[10] = q2[2]; r[11] = q2[3];
  r[12] = q3[0]; r[13] = q3[1]; r[14] = q3[2]; r[15] = q3[3];
  return r;
}

// B-matrix 32x16: same K order as A: halves 0..7 hold K = hi*8..hi*8+7, halves 8..15 hold K = 16+hi*8..16+hi*8+7.
__device__ __forceinline__ bf16x16 load_b_k32(const bf16* __restrict__ p, int hi) {
  bf16x8 lo = *(const bf16x8*)(p + hi * 8);
  bf16x8 hi8 = *(const bf16x8*)(p + 16 + hi * 8);
  bf16x16 r;
#pragma unroll
  for (int i = 0; i < 8; ++i) { r[i] = lo[i]; r[8 + i] = hi8[i]; }
  return r;
}

__device__ __forceinline__ float fast_silu(float h) {
  // h * rcp(1 + exp(-h)) : one v_exp_f32 + one v_rcp_f32 instead of IEEE div.
  return h * __builtin_amdgcn_rcpf(1.0f + __expf(-h));
}

// ---------------------------------------------------------------------------
// Kernel 0: zero the output accumulator and per-expert counters.
// ---------------------------------------------------------------------------
__global__ void moe_zero_kernel(float* __restrict__ out, int n, int* __restrict__ counts) {
  int i = blockIdx.x * blockDim.x + threadIdx.x;
  if (i < EE) counts[i] = 0;
  for (; i < n; i += gridDim.x * blockDim.x) out[i] = 0.0f;
}

// ---------------------------------------------------------------------------
// Kernel 1: stream fp32 -> bf16 weight conversion (4-wide).
// ---------------------------------------------------------------------------
__global__ void moe_cvt_kernel(const float* __restrict__ in, bf16* __restrict__ out, int n4) {
  int i = blockIdx.x * blockDim.x + threadIdx.x;
  int stride = gridDim.x * blockDim.x;
  for (; i < n4; i += stride) {
    f32x4 v = ((const f32x4*)in)[i];
    bf16x4 o;
    o[0] = (bf16)v[0]; o[1] = (bf16)v[1]; o[2] = (bf16)v[2]; o[3] = (bf16)v[3];
    ((bf16x4*)out)[i] = o;
  }
}

// ---------------------------------------------------------------------------
// Kernel 2: router. One wave32 per token: logits, softmax, top-2,
// softmax over the top-2 probs (as reference), scatter into expert lists.
// ---------------------------------------------------------------------------
__global__ __launch_bounds__(256)
void moe_router_kernel(const float* __restrict__ x, const float* __restrict__ rw,
                       const float* __restrict__ escale,
                       int* __restrict__ counts, int* __restrict__ tok,
                       float* __restrict__ wgt) {
  int lane = threadIdx.x & 31;
  int wv = threadIdx.x >> 5;
  int t = blockIdx.x * (blockDim.x >> 5) + wv;
  if (t >= TT) return;

  float acc[EE];
#pragma unroll
  for (int e = 0; e < EE; ++e) acc[e] = 0.0f;

  const float* xr = x + (size_t)t * DD;
  for (int d = lane; d < DD; d += 32) {
    float xv = xr[d];
#pragma unroll
    for (int e = 0; e < EE; ++e) acc[e] = fmaf(xv, rw[e * DD + d], acc[e]);
  }
#pragma unroll
  for (int e = 0; e < EE; ++e) {
    for (int off = 16; off > 0; off >>= 1) acc[e] += __shfl_xor(acc[e], off, 32);
  }

  if (lane == 0) {
    float mx = acc[0];
#pragma unroll
    for (int e = 1; e < EE; ++e) mx = fmaxf(mx, acc[e]);
    float p[EE];
    float s = 0.0f;
#pragma unroll
    for (int e = 0; e < EE; ++e) { p[e] = __expf(acc[e] - mx); s += p[e]; }
    float inv = __builtin_amdgcn_rcpf(s);
#pragma unroll
    for (int e = 0; e < EE; ++e) p[e] *= inv;

    int i0 = 0;
#pragma unroll
    for (int e = 1; e < EE; ++e) if (p[e] > p[i0]) i0 = e;
    int i1 = (i0 == 0) ? 1 : 0;
#pragma unroll
    for (int e = 0; e < EE; ++e) if (e != i0 && p[e] > p[i1]) i1 = e;

    // Reference: second softmax over the two selected probabilities.
    float w0 = 1.0f / (1.0f + __expf(p[i1] - p[i0]));
    float w1 = 1.0f - w0;

    int pos0 = atomicAdd(&counts[i0], 1);
    tok[i0 * TT + pos0] = t;
    wgt[i0 * TT + pos0] = w0 * escale[i0];
    int pos1 = atomicAdd(&counts[i1], 1);
    tok[i1 * TT + pos1] = t;
    wgt[i1 * TT + pos1] = w1 * escale[i1];
  }
}

// ---------------------------------------------------------------------------
// Kernel 3: fused grouped expert MLP. One workgroup = 16 tokens of one expert.
// 8 waves; Y accumulators live in registers across F-chunks. K-loop is the
// outer loop per phase so each A fragment is loaded once and fed to 4 (phase
// A) / 8 (phase B) back-to-back WMMAs against different B fragments.
// ---------------------------------------------------------------------------
__global__ __launch_bounds__(256)
void moe_mlp_kernel(const float* __restrict__ x,
                    const bf16* __restrict__ w1b, const bf16* __restrict__ w2b,
                    const int* __restrict__ counts, const int* __restrict__ tok,
                    const float* __restrict__ wgt, float* __restrict__ out) {
  __shared__ bf16 ldsX[16 * DD];        // 32 KB: gathered token tile (bf16)
  __shared__ bf16 ldsH[16 * CHUNK_F];   // 16 KB: SiLU(h) chunk
  __shared__ int   sTok[16];
  __shared__ float sWgt[16];

  int e = blockIdx.x >> 7;              // 128 token-tiles per expert
  int mt = blockIdx.x & 127;
  int n = counts[e];
  int row0 = mt * 16;
  if (row0 >= n) return;                // block-uniform early exit

  int tid = threadIdx.x;
  int lane = tid & 31;
  int wv = tid >> 5;
  int hi = (lane >> 4) & 1;
  int ln = lane & 15;
  int kofs = hi * 8;

  if (tid < 16) {
    int slot = row0 + tid;
    sTok[tid] = (slot < n) ? tok[e * TT + slot] : -1;
    sWgt[tid] = (slot < n) ? wgt[e * TT + slot] : 0.0f;
  }

  // Stage gathered X rows into LDS with fp32->bf16 conversion.
  {
    int r = tid >> 4;                   // 16 rows
    int c0 = (tid & 15) * 64;           // 64 contiguous elems per thread
    int slot = row0 + r;
    int tt = (slot < n) ? tok[e * TT + slot] : -1;
    if (tt >= 0) {
      const f32x4* src = (const f32x4*)(x + (size_t)tt * DD + c0);
#pragma unroll
      for (int i = 0; i < 16; ++i) {
        f32x4 v = src[i];
        bf16x4 o;
        o[0] = (bf16)v[0]; o[1] = (bf16)v[1]; o[2] = (bf16)v[2]; o[3] = (bf16)v[3];
        *(bf16x4*)&ldsX[r * DD + c0 + i * 4] = o;
      }
    } else {
#pragma unroll
      for (int i = 0; i < 16; ++i) {
        bf16x4 o;
        o[0] = (bf16)0.f; o[1] = (bf16)0.f; o[2] = (bf16)0.f; o[3] = (bf16)0.f;
        *(bf16x4*)&ldsX[r * DD + c0 + i * 4] = o;
      }
    }
  }
  __syncthreads();

  // Per-wave Y accumulators: wave wv owns d-tiles d0 = wv*128 + j*16.
  f32x8 accY[8];
#pragma unroll
  for (int j = 0; j < 8; ++j) accY[j] = zero8();

  for (int fc = 0; fc < FF; fc += CHUNK_F) {
    // ---- Phase A: H[:, fc:fc+512] = silu(X @ W1^T) ----
    // Wave handles 4 f-tiles; A loaded once per K-step, 4 WMMAs per step.
    {
      f32x8 accH[4];
#pragma unroll
      for (int j = 0; j < 4; ++j) accH[j] = zero8();

      const bf16* ap = &ldsX[ln * DD];
      const bf16* bp0 = w1b + ((size_t)e * FF + (size_t)(fc + wv * 64 + ln)) * DD;
#pragma unroll 2
      for (int k0 = 0; k0 < DD; k0 += 32) {
        bf16x16 a = load_a_k32(ap + k0, kofs);
#pragma unroll
        for (int j = 0; j < 4; ++j) {
          bf16x16 b = load_b_k32(bp0 + (size_t)(j * 16) * DD + k0, hi);
          accH[j] = __builtin_amdgcn_wmma_f32_16x16x32_bf16(false, a, false, b,
                                                            (short)0, accH[j],
                                                            false, false);
        }
      }
      // SiLU + store bf16 H chunk to LDS.
#pragma unroll
      for (int j = 0; j < 4; ++j) {
        int fl = wv * 64 + j * 16;
#pragma unroll
        for (int v = 0; v < 8; ++v) {
          int m = v + hi * 8;
          ldsH[m * CHUNK_F + fl + ln] = (bf16)fast_silu(accH[j][v]);
        }
      }
    }
    __syncthreads();

    // ---- Phase B: Y += H_chunk @ W2_chunk^T ----
    // Wave handles 8 d-tiles; A loaded once per K-step, 8 WMMAs per step.
    {
      const bf16* ap = &ldsH[ln * CHUNK_F];
      const bf16* bp0 = w2b + ((size_t)e * DD + (size_t)(wv * 128 + ln)) * FF + fc;
#pragma unroll 1
      for (int k0 = 0; k0 < CHUNK_F; k0 += 32) {
        bf16x16 a = load_a_k32(ap + k0, kofs);
#pragma unroll
        for (int j = 0; j < 8; ++j) {
          bf16x16 b = load_b_k32(bp0 + (size_t)(j * 16) * FF + k0, hi);
          accY[j] = __builtin_amdgcn_wmma_f32_16x16x32_bf16(false, a, false, b,
                                                            (short)0, accY[j],
                                                            false, false);
        }
      }
    }
    __syncthreads();
  }

  // ---- Epilogue: weighted scatter-add into out[t, :] ----
#pragma unroll
  for (int j = 0; j < 8; ++j) {
    int d0 = wv * 128 + j * 16;
#pragma unroll
    for (int v = 0; v < 8; ++v) {
      int m = v + hi * 8;
      int tt = sTok[m];
      if (tt >= 0) {
        atomicAdd(&out[(size_t)tt * DD + (size_t)(d0 + ln)], accY[j][v] * sWgt[m]);
      }
    }
  }
}

// ---------------------------------------------------------------------------
// Host launcher.
// ---------------------------------------------------------------------------
extern "C" void kernel_launch(void* const* d_in, const int* in_sizes, int n_in,
                              void* d_out, int out_size, void* d_ws, size_t ws_size,
                              hipStream_t stream) {
  const float* x   = (const float*)d_in[0];   // [B,S,D]
  const float* rw  = (const float*)d_in[1];   // [E,D]
  const float* w1  = (const float*)d_in[2];   // [E,F,D]
  const float* w2  = (const float*)d_in[3];   // [E,D,F]
  const float* esc = (const float*)d_in[4];   // [E]
  float* out = (float*)d_out;                 // [B,S,D] fp32

  // Workspace layout:
  //   [0,256)                      : counts (E ints, padded)
  //   [256, 256+E*T*4)             : token lists
  //   [+E*T*4, +E*T*8)             : slot weights (incl. expert_scale)
  //   then w1 bf16 (64 MB), w2 bf16 (64 MB)
  char* ws = (char*)d_ws;
  int*   counts = (int*)ws;
  int*   tokL   = (int*)(ws + 256);
  float* wgtL   = (float*)(ws + 256 + (size_t)EE * TT * 4);
  bf16*  w1b    = (bf16*)(ws + 256 + (size_t)EE * TT * 8);
  bf16*  w2b    = w1b + (size_t)EE * FF * DD;

  const int nOut = TT * DD;
  moe_zero_kernel<<<512, 256, 0, stream>>>(out, nOut, counts);

  const int n4 = (EE * FF * DD) / 4;
  moe_cvt_kernel<<<4096, 256, 0, stream>>>(w1, w1b, n4);
  moe_cvt_kernel<<<4096, 256, 0, stream>>>(w2, w2b, n4);

  moe_router_kernel<<<TT / 8, 256, 0, stream>>>(x, rw, esc, counts, tokL, wgtL);

  moe_mlp_kernel<<<EE * 128, 256, 0, stream>>>(x, w1b, w2b, counts, tokL, wgtL, out);
}
